// ClinicalDependencyEncoder_48155173323222
// MI455X (gfx1250) — compile-verified, matrix operand order checked by reading, not run on hardware
//
#include <hip/hip_runtime.h>

#define N_NODES 6144
#define EMB     256
#define NHEADS  4
#define HDIM    64

typedef unsigned short u16;
typedef unsigned int   u32;
typedef __attribute__((ext_vector_type(16))) __bf16 v16bf;
typedef __attribute__((ext_vector_type(8)))  float  v8f;

union ABTile { v16bf v; u32 u[8]; };

__device__ __forceinline__ u16 f32_to_bf16(float f) {
  u32 u = __float_as_uint(f);
  u32 r = (u + 0x7FFFu + ((u >> 16) & 1u)) >> 16;   // round-to-nearest-even
  return (u16)r;
}

// Load one 16x32 bf16 WMMA A/B tile fragment for this lane.
// Layout (ISA 7.12.2, 16-bit A 16x32): lane L holds row/col (L&15); element e
// maps to k = e + 8*(e>>3) + 8*(L>>4)  -> dword gathers at {0,1,2,3,8,9,10,11}+4*half.
// (Compiler merges these into two global_load_b128 per tile.)
__device__ __forceinline__ v16bf load_tile_row(const u16* rowptr, int kbase, int half) {
  ABTile t;
  const u32* p = (const u32*)(rowptr + kbase + 8 * half);
  t.u[0] = p[0];  t.u[1] = p[1];  t.u[2] = p[2];  t.u[3] = p[3];
  t.u[4] = p[8];  t.u[5] = p[9];  t.u[6] = p[10]; t.u[7] = p[11];
  return t.v;
}

__device__ __forceinline__ v8f wmma_bf16(v16bf a, v16bf b, v8f c) {
  return __builtin_amdgcn_wmma_f32_16x16x32_bf16(false, a, false, b, (short)0, c,
                                                 false, false);
}

__device__ __forceinline__ float halfmax16(float v) {
  v = fmaxf(v, __shfl_xor(v, 1));
  v = fmaxf(v, __shfl_xor(v, 2));
  v = fmaxf(v, __shfl_xor(v, 4));
  v = fmaxf(v, __shfl_xor(v, 8));
  return v;
}
__device__ __forceinline__ float halfsum16(float v) {
  v += __shfl_xor(v, 1);
  v += __shfl_xor(v, 2);
  v += __shfl_xor(v, 4);
  v += __shfl_xor(v, 8);
  return v;
}

// ---------------------------------------------------------------- f32 -> bf16
__global__ void cvt_bf16_kernel(const float* __restrict__ src, u16* __restrict__ dst, int n) {
  int i = blockIdx.x * blockDim.x + threadIdx.x;
  if (i < n) dst[i] = f32_to_bf16(src[i]);
}

// ------------------------------------------------- Y = bf16(Xb @ W^T + bias)
// 4 waves / block, one 16x16 output tile per wave.
// transpose_out=1 stores Y transposed ([EMB][N]) for V (feature-major layout).
__global__ void proj_kernel(const u16* __restrict__ xb, const u16* __restrict__ wb,
                            const float* __restrict__ bias, u16* __restrict__ out,
                            int transpose_out) {
  int wave = threadIdx.x >> 5;
  int lane = threadIdx.x & 31;
  int tile = blockIdx.x * 4 + wave;           // 0 .. (N/16)*(E/16)-1
  int jt = tile & 15;                         // E/16 = 16 column tiles
  int nt = tile >> 4;                         // row tile
  int row0 = nt * 16, col0 = jt * 16;
  int half = lane >> 4, l15 = lane & 15;

  const u16* arow = xb + (size_t)(row0 + l15) * EMB;   // A: node rows
  const u16* brow = wb + (size_t)(col0 + l15) * EMB;   // B: W row j = output col j
  v8f acc = {};
#pragma unroll
  for (int k = 0; k < EMB; k += 32) {
    v16bf a = load_tile_row(arow, k, half);
    v16bf b = load_tile_row(brow, k, half);
    acc = wmma_bf16(a, b, acc);
  }
  float bval = bias[col0 + l15];
#pragma unroll
  for (int r = 0; r < 8; ++r) {
    int m = r + 8 * half;
    u16 bf = f32_to_bf16(acc[r] + bval);
    if (!transpose_out) out[(size_t)(row0 + m) * EMB + col0 + l15] = bf;
    else                out[(size_t)(col0 + l15) * N_NODES + row0 + m] = bf;
  }
}

// -------------------------------------------- masked flash attention
// One block = one 16-query tile x ALL 4 heads (wave w = head w).
// The 16x32 adjacency tile is loaded once per block into LDS (coalesced
// float4) and shared by all heads -> adjacency HBM traffic hits the 151 MB
// floor instead of 4x that. 32 keys per iteration.
__global__ void attn_kernel(const u16* __restrict__ Qb, const u16* __restrict__ Kb,
                            const u16* __restrict__ Vt, const float* __restrict__ adj,
                            u16* __restrict__ Ob) {
  __shared__ float adjS[16][32];              // shared adjacency tile (2 KB)
  __shared__ u16 P[NHEADS][16][32];           // per-head P tile (16q x 32key bf16)
  int h    = threadIdx.x >> 5;                // wave index == head
  int lane = threadIdx.x & 31;
  int q0   = blockIdx.x * 16;
  int half = lane >> 4, l15 = lane & 15;
  int arow_t = threadIdx.x >> 3;              // adj loader: 16 rows x 8 float4 segs
  int aseg_t = threadIdx.x & 7;
  const float scale = 0.125f;                 // 1/sqrt(64)

  // Q fragments live in registers for all 192 key iterations
  const u16* qrow = Qb + (size_t)(q0 + l15) * EMB + h * HDIM;
  v16bf qa0 = load_tile_row(qrow, 0,  half);
  v16bf qa1 = load_tile_row(qrow, 32, half);

  float rmax[8], rsum[8];
  v8f o[4];
#pragma unroll
  for (int r = 0; r < 8; ++r) { rmax[r] = -1e30f; rsum[r] = 0.0f; }
#pragma unroll
  for (int t = 0; t < 4; ++t) o[t] = (v8f){};

  for (int kb = 0; kb < N_NODES; kb += 32) {
    // cooperative, coalesced adjacency tile load (one float4 per thread)
    const float* aptr = adj + (size_t)(q0 + arow_t) * N_NODES + kb + aseg_t * 4;
    float4 a4 = *(const float4*)aptr;
    *(float4*)&adjS[arow_t][aseg_t * 4] = a4;
    if (kb + 32 < N_NODES)                     // gfx1250 global_prefetch path
      __builtin_prefetch(aptr + 32, 0, 1);
    __syncthreads();

    // S = Q K^T for two 16-key subtiles (head h)
    const u16* krow0 = Kb + (size_t)(kb + l15) * EMB + h * HDIM;
    const u16* krow1 = Kb + (size_t)(kb + 16 + l15) * EMB + h * HDIM;
    v8f s0 = {}, s1 = {};
    s0 = wmma_bf16(qa0, load_tile_row(krow0, 0,  half), s0);
    s0 = wmma_bf16(qa1, load_tile_row(krow0, 32, half), s0);
    s1 = wmma_bf16(qa0, load_tile_row(krow1, 0,  half), s1);
    s1 = wmma_bf16(qa1, load_tile_row(krow1, 32, half), s1);

#pragma unroll
    for (int r = 0; r < 8; ++r) {
      int m = r + 8 * half;
      float a0 = adjS[m][l15];
      float a1 = adjS[m][16 + l15];
      float v0 = (a0 == 0.0f) ? -3e38f : s0[r] * scale;
      float v1 = (a1 == 0.0f) ? -3e38f : s1[r] * scale;

      // online softmax update for row m (16 lanes of this half share the row)
      float tilemax = halfmax16(fmaxf(v0, v1));
      float nm   = fmaxf(rmax[r], tilemax);
      float corr = __expf(rmax[r] - nm);
      float p0   = __expf(v0 - nm);            // masked -> underflow to 0
      float p1   = __expf(v1 - nm);
      rsum[r] = rsum[r] * corr + halfsum16(p0 + p1);
      rmax[r] = nm;
#pragma unroll
      for (int t = 0; t < 4; ++t) o[t][r] *= corr;

      P[h][m][l15]      = f32_to_bf16(p0);
      P[h][m][16 + l15] = f32_to_bf16(p1);
    }
    __syncthreads();   // P visible; adjS reads complete (safe to overwrite next iter)

    // P (A-layout via LDS bounce) x V^T tiles
    ABTile pa;
    const u32* pr = (const u32*)&P[h][l15][0];
    int pb = 4 * half;
    pa.u[0] = pr[pb + 0]; pa.u[1] = pr[pb + 1]; pa.u[2] = pr[pb + 2]; pa.u[3] = pr[pb + 3];
    pa.u[4] = pr[pb + 8]; pa.u[5] = pr[pb + 9]; pa.u[6] = pr[pb + 10]; pa.u[7] = pr[pb + 11];
#pragma unroll
    for (int t = 0; t < 4; ++t) {
      const u16* vrow = Vt + (size_t)(h * HDIM + t * 16 + l15) * N_NODES;
      v16bf vb = load_tile_row(vrow, kb, half);
      o[t] = wmma_bf16(pa.v, vb, o[t]);
    }
    // P reads finish before next softmax overwrites P (next iter's first sync)
  }

  // normalize and store attention output (bf16, row-major N x E)
#pragma unroll
  for (int t = 0; t < 4; ++t) {
#pragma unroll
    for (int r = 0; r < 8; ++r) {
      int m = r + 8 * half;
      float val = o[t][r] / rsum[r];
      Ob[(size_t)(q0 + m) * EMB + h * HDIM + t * 16 + l15] = f32_to_bf16(val);
    }
  }
}

// ------------------- Y = Ob @ Wo^T + bo + x, then LayerNorm; 16 rows / block
__global__ void outproj_ln_kernel(const u16* __restrict__ Ob, const u16* __restrict__ Wob,
                                  const float* __restrict__ bo, const float* __restrict__ x,
                                  const float* __restrict__ gamma, const float* __restrict__ beta,
                                  float* __restrict__ y) {
  __shared__ float yb[16][EMB];
  __shared__ float mu[16], rstd[16];
  int wave = threadIdx.x >> 5;
  int lane = threadIdx.x & 31;
  int half = lane >> 4, l15 = lane & 15;
  int row0 = blockIdx.x * 16;
  int j0 = wave * 64;                          // each wave: 4 column tiles

  const u16* arow = Ob + (size_t)(row0 + l15) * EMB;
  v8f acc[4];
#pragma unroll
  for (int t = 0; t < 4; ++t) acc[t] = (v8f){};
#pragma unroll
  for (int k = 0; k < EMB; k += 32) {
    v16bf a = load_tile_row(arow, k, half);
#pragma unroll
    for (int t = 0; t < 4; ++t) {
      const u16* brow = Wob + (size_t)(j0 + t * 16 + l15) * EMB;
      v16bf b = load_tile_row(brow, k, half);
      acc[t] = wmma_bf16(a, b, acc[t]);
    }
  }
#pragma unroll
  for (int t = 0; t < 4; ++t) {
    int col = j0 + t * 16 + l15;
#pragma unroll
    for (int r = 0; r < 8; ++r) {
      int m = r + 8 * half;
      yb[m][col] = acc[t][r] + bo[col] + x[(size_t)(row0 + m) * EMB + col];
    }
  }
  __syncthreads();

  if (threadIdx.x < 16) {
    int m = threadIdx.x;
    float s = 0.0f, s2 = 0.0f;
    for (int c = 0; c < EMB; ++c) { float v = yb[m][c]; s += v; s2 += v * v; }
    float mean = s * (1.0f / EMB);
    float var  = s2 * (1.0f / EMB) - mean * mean;
    mu[m] = mean;
    rstd[m] = rsqrtf(var + 1e-5f);
  }
  __syncthreads();

  for (int i = threadIdx.x; i < 16 * EMB; i += blockDim.x) {
    int m = i >> 8, c = i & 255;
    y[(size_t)(row0 + m) * EMB + c] = (yb[m][c] - mu[m]) * rstd[m] * gamma[c] + beta[c];
  }
}

// ---------------------------------------------------------------------------
extern "C" void kernel_launch(void* const* d_in, const int* in_sizes, int n_in,
                              void* d_out, int out_size, void* d_ws, size_t ws_size,
                              hipStream_t stream) {
  const float* x     = (const float*)d_in[0];
  const float* adj   = (const float*)d_in[1];
  const float* Wq    = (const float*)d_in[2];
  const float* bq    = (const float*)d_in[3];
  const float* Wk    = (const float*)d_in[4];
  const float* bk    = (const float*)d_in[5];
  const float* Wv    = (const float*)d_in[6];
  const float* bv    = (const float*)d_in[7];
  const float* Wo    = (const float*)d_in[8];
  const float* bo    = (const float*)d_in[9];
  const float* gamma = (const float*)d_in[10];
  const float* beta  = (const float*)d_in[11];
  float* y = (float*)d_out;

  char* ws = (char*)d_ws;
  size_t off = 0;
  const size_t NE = (size_t)N_NODES * EMB * sizeof(u16);   // 3 MB each
  const size_t EE = (size_t)EMB * EMB * sizeof(u16);       // 128 KB each
  u16* xb  = (u16*)(ws + off); off += NE;
  u16* Qb  = (u16*)(ws + off); off += NE;
  u16* Kb  = (u16*)(ws + off); off += NE;
  u16* Vt  = (u16*)(ws + off); off += NE;   // [EMB][N] feature-major
  u16* Ob  = (u16*)(ws + off); off += NE;
  u16* Wqb = (u16*)(ws + off); off += EE;
  u16* Wkb = (u16*)(ws + off); off += EE;
  u16* Wvb = (u16*)(ws + off); off += EE;
  u16* Wob = (u16*)(ws + off); off += EE;
  (void)ws_size; (void)in_sizes; (void)n_in; (void)out_size;

  int ne = N_NODES * EMB, ee = EMB * EMB;
  cvt_bf16_kernel<<<(ne + 255) / 256, 256, 0, stream>>>(x,  xb,  ne);
  cvt_bf16_kernel<<<(ee + 255) / 256, 256, 0, stream>>>(Wq, Wqb, ee);
  cvt_bf16_kernel<<<(ee + 255) / 256, 256, 0, stream>>>(Wk, Wkb, ee);
  cvt_bf16_kernel<<<(ee + 255) / 256, 256, 0, stream>>>(Wv, Wvb, ee);
  cvt_bf16_kernel<<<(ee + 255) / 256, 256, 0, stream>>>(Wo, Wob, ee);

  int ntiles = (N_NODES / 16) * (EMB / 16);                // 6144 tiles, 4/block
  proj_kernel<<<ntiles / 4, 128, 0, stream>>>(xb, Wqb, bq, Qb, 0);
  proj_kernel<<<ntiles / 4, 128, 0, stream>>>(xb, Wkb, bk, Kb, 0);
  proj_kernel<<<ntiles / 4, 128, 0, stream>>>(xb, Wvb, bv, Vt, 1);

  // one block per 16-query tile, all 4 heads inside (adj read exactly once)
  attn_kernel<<<N_NODES / 16, 32 * NHEADS, 0, stream>>>(Qb, Kb, Vt, adj, Ob);

  outproj_ln_kernel<<<N_NODES / 16, 128, 0, stream>>>(Ob, Wob, bo, x, gamma, beta, y);
}
